// MultiLabelAngleProtoLoss_75015898792474
// MI455X (gfx1250) — matrix operand order repaired and checked by reading, not compiled, not run on hardware
//
#include <hip/hip_runtime.h>

typedef __attribute__((ext_vector_type(16))) _Float16 v16h;
typedef __attribute__((ext_vector_type(8)))  _Float16 v8h;
typedef __attribute__((ext_vector_type(2)))  _Float16 v2h;
typedef __attribute__((ext_vector_type(8)))  float    v8f;

constexpr int N_ = 65536, D_ = 512, C_ = 256;

// ---- workspace layout (bytes) ----
constexpr size_t WS_P      = 0;                        // C*D f32 accumulation of label^T @ x
constexpr size_t WS_COUNTS = (size_t)C_ * D_ * 4;      // C f32
constexpr size_t WS_ACTIVE = WS_COUNTS + C_ * 4;       // C f32
constexpr size_t WS_MISC   = WS_ACTIVE + C_ * 4;       // [0]=n_active, [1]=loss_sum
constexpr size_t WS_AH     = WS_MISC + 64;             // C*D f16, 32B aligned (526400 % 32 == 0)
constexpr int    WS_ZERO_FLOATS = (int)(WS_AH / 4);    // zero P, counts, active, misc

static __device__ __forceinline__ v16h cat8(v8h lo, v8h hi) {
  return __builtin_shufflevector(lo, hi, 0, 1, 2, 3, 4, 5, 6, 7,
                                 8, 9, 10, 11, 12, 13, 14, 15);
}
static __device__ __forceinline__ v2h pk2(float a, float b) {
  v2h r; r[0] = (_Float16)a; r[1] = (_Float16)b; return r;   // -> v_cvt_pk_f16_f32
}

// ---------------- init: zero accumulators ----------------
__global__ void k_init(float* ws) {
  int i = blockIdx.x * blockDim.x + threadIdx.x;
  if (i < WS_ZERO_FLOATS) ws[i] = 0.f;
}

// ---------------- counts[c] = sum_n label[n][c] ----------------
__global__ void __launch_bounds__(256) k_counts(const int* __restrict__ label,
                                                float* __restrict__ counts) {
  const int c  = threadIdx.x;          // 256 threads == C
  const int r0 = blockIdx.x * 256;     // 256 blocks of 256 rows
  int s = 0;
  for (int r = 0; r < 256; ++r) s += label[(size_t)(r0 + r) * C_ + c];
  atomicAdd(&counts[c], (float)s);
}

// ---------------- GEMM1: P[c][d] += sum_n label[n][c] * x[n][d] ----------------
// grid = (64 n-chunks of 1024, 4 d-groups of 128, 4 c-groups of 64), block = 256 (8 waves)
// LDS-staged: transposed f16 tiles, +8-half padding (row stride 80B); n-pair packed b32 stores
__global__ void __launch_bounds__(256) k_gemm1(const float* __restrict__ x,
                                               const int* __restrict__ label,
                                               float* __restrict__ P) {
  __shared__ __align__(16) _Float16 labT[64][40];   // [c][n + pad]
  __shared__ __align__(16) _Float16 xT[128][40];    // [d][n + pad]

  const int t     = threadIdx.x;
  const int lane  = t & 31;
  const int wave  = t >> 5;
  const int ln16  = lane & 15;
  const int kbase = (lane < 16) ? 0 : 8;   // A-frag K offset per half-wave
  const int khalf = lane >> 4;             // B-frag K half per half-wave
  const int nbase = blockIdx.x * 1024;
  const int dbase = blockIdx.y * 128;
  const int cbase = blockIdx.z * 64;
  const int ctl   = wave >> 1;             // local ctile 0..3
  const int dqu   = wave & 1;              // dtiles [dqu*4, dqu*4+4)

  v8f acc[4];
  for (int j = 0; j < 4; ++j)
    for (int v = 0; v < 8; ++v) acc[j][v] = 0.f;

  const int c_l = ctl * 16 + ln16;
  // staging coordinates
  const int lnp = (t >> 4) * 2;        // label: n-pair 0,2,..,30
  const int lc4 = (t & 15) * 4;        // label: c quad
  const int xd4 = (t & 31) * 4;        // x: d quad
  const int xnp = (t >> 5) * 2;        // x: n-pair base (+16 on pass 1)

  for (int it = 0; it < 32; ++it) {
    const int n0 = nbase + it * 32;
    __syncthreads();  // previous iteration's readers done before overwrite
    // stage label tile [32 n][64 c] -> labT[c][n], packed n-pairs (ds_store_b32)
    {
      const int4 la = *(const int4*)&label[(size_t)(n0 + lnp)     * C_ + cbase + lc4];
      const int4 lb = *(const int4*)&label[(size_t)(n0 + lnp + 1) * C_ + cbase + lc4];
      *(v2h*)&labT[lc4 + 0][lnp] = pk2((float)la.x, (float)lb.x);
      *(v2h*)&labT[lc4 + 1][lnp] = pk2((float)la.y, (float)lb.y);
      *(v2h*)&labT[lc4 + 2][lnp] = pk2((float)la.z, (float)lb.z);
      *(v2h*)&labT[lc4 + 3][lnp] = pk2((float)la.w, (float)lb.w);
      if (it + 1 < 32)
        __builtin_prefetch(&label[(size_t)(n0 + 32 + lnp) * C_ + cbase + lc4], 0, 3);
    }
    // stage x tile [32 n][128 d] -> xT[d][n], packed n-pairs (ds_store_b32)
#pragma unroll
    for (int p = 0; p < 2; ++p) {
      const int n = xnp + p * 16;
      const float4 xa = *(const float4*)&x[(size_t)(n0 + n)     * D_ + dbase + xd4];
      const float4 xb = *(const float4*)&x[(size_t)(n0 + n + 1) * D_ + dbase + xd4];
      *(v2h*)&xT[xd4 + 0][n] = pk2(xa.x, xb.x);
      *(v2h*)&xT[xd4 + 1][n] = pk2(xa.y, xb.y);
      *(v2h*)&xT[xd4 + 2][n] = pk2(xa.z, xb.z);
      *(v2h*)&xT[xd4 + 3][n] = pk2(xa.w, xb.w);
      if (p == 0 && it + 1 < 32)
        __builtin_prefetch(&x[(size_t)(n0 + 32 + n) * D_ + dbase + xd4], 0, 3);
    }
    __syncthreads();

    // A-fragment: A[M=c_local][K=n] = labT[c_l][k], two 8-half runs (2x ds_load_b128)
    const v8h alo = *(const v8h*)&labT[c_l][kbase];
    const v8h ahi = *(const v8h*)&labT[c_l][16 + kbase];
    const v16h a  = cat8(alo, ahi);
#pragma unroll
    for (int j = 0; j < 4; ++j) {
      const int d_l = dqu * 64 + j * 16 + ln16;
      // B-fragment: B[K=n][N=d] = xT[d_l][k], two 8-half runs matching A
      const v8h blo = *(const v8h*)&xT[d_l][kbase];
      const v8h bhi = *(const v8h*)&xT[d_l][16 + kbase];
      acc[j] = __builtin_amdgcn_wmma_f32_16x16x32_f16(false, a, false, cat8(blo, bhi),
                                                      (short)0, acc[j], false, false);
    }
  }

  // C/D layout: lanes 0-15: N=lane, M=v ; lanes 16-31: N=lane-16, M=v+8
  const int mo = (lane < 16) ? 0 : 8;
#pragma unroll
  for (int j = 0; j < 4; ++j) {
    const int d = dbase + dqu * 64 + j * 16 + ln16;
#pragma unroll
    for (int v = 0; v < 8; ++v)
      atomicAdd(&P[(size_t)(cbase + ctl * 16 + v + mo) * D_ + d], acc[j][v]);
  }
}

// ---------------- finalize anchors: divide by count, normalize rows, emit f16 ----------------
__global__ void __launch_bounds__(256) k_finalize(const float* __restrict__ P,
                                                  const float* __restrict__ counts,
                                                  float* __restrict__ active,
                                                  float* __restrict__ misc,
                                                  _Float16* __restrict__ Ah) {
  const int c = blockIdx.x;
  const int t = threadIdx.x;
  const float cnt = counts[c];
  const float inv = 1.f / fmaxf(cnt, 1.f);
  const float a0 = P[(size_t)c * D_ + t]       * inv;
  const float a1 = P[(size_t)c * D_ + t + 256] * inv;
  __shared__ float sm[256];
  sm[t] = a0 * a0 + a1 * a1;
  __syncthreads();
  for (int s = 128; s > 0; s >>= 1) {
    if (t < s) sm[t] += sm[t + s];
    __syncthreads();
  }
  __shared__ float rnorm_s;
  if (t == 0) {
    const float nrm = sqrtf(sm[0]);
    rnorm_s = (nrm > 0.f) ? 1.f / nrm : 0.f;
    active[c] = (cnt > 0.f) ? 1.f : 0.f;
    if (cnt > 0.f) atomicAdd(&misc[0], 1.f);
  }
  __syncthreads();
  const float r = rnorm_s;
  Ah[(size_t)c * D_ + t]       = (_Float16)(a0 * r);
  Ah[(size_t)c * D_ + t + 256] = (_Float16)(a1 * r);
}

// ---------------- GEMM2 + fused BCE loss ----------------
// grid = N/64 blocks; block = 256 (8 waves); wave w: mtile = w>>1, ntile base = (w&1)*8
__global__ void __launch_bounds__(256) k_gemm2_loss(const float* __restrict__ x,
                                                    const int* __restrict__ label,
                                                    const _Float16* __restrict__ Ah,
                                                    const float* __restrict__ active,
                                                    const float* __restrict__ wp,
                                                    const float* __restrict__ bp,
                                                    float* __restrict__ misc) {
  const int lane  = threadIdx.x & 31;
  const int wave  = threadIdx.x >> 5;
  const int ln16  = lane & 15;
  const int kbase = (lane < 16) ? 0 : 8;
  const int khalf = lane >> 4;
  const int m0    = blockIdx.x * 64;
  const int mtile = wave >> 1;
  const int ntb   = (wave & 1) * 8;
  const int m     = m0 + mtile * 16 + ln16;

  v8f acc[8];
  for (int j = 0; j < 8; ++j)
    for (int v = 0; v < 8; ++v) acc[j][v] = 0.f;
  float sq = 0.f;

  const float* xrow = x + (size_t)m * D_;
  for (int kt = 0; kt < 16; ++kt) {
    const int k0 = kt * 32;
    // A-fragment: two contiguous 8-float runs per lane -> 4x global_load_b128
    const float4 r0 = *(const float4*)&xrow[k0 + kbase];
    const float4 r1 = *(const float4*)&xrow[k0 + kbase + 4];
    const float4 r2 = *(const float4*)&xrow[k0 + 16 + kbase];
    const float4 r3 = *(const float4*)&xrow[k0 + 16 + kbase + 4];
    sq += r0.x * r0.x + r0.y * r0.y + r0.z * r0.z + r0.w * r0.w +
          r1.x * r1.x + r1.y * r1.y + r1.z * r1.z + r1.w * r1.w +
          r2.x * r2.x + r2.y * r2.y + r2.z * r2.z + r2.w * r2.w +
          r3.x * r3.x + r3.y * r3.y + r3.z * r3.z + r3.w * r3.w;
    v16h a;
    a[0]  = (_Float16)r0.x; a[1]  = (_Float16)r0.y; a[2]  = (_Float16)r0.z; a[3]  = (_Float16)r0.w;
    a[4]  = (_Float16)r1.x; a[5]  = (_Float16)r1.y; a[6]  = (_Float16)r1.z; a[7]  = (_Float16)r1.w;
    a[8]  = (_Float16)r2.x; a[9]  = (_Float16)r2.y; a[10] = (_Float16)r2.z; a[11] = (_Float16)r2.w;
    a[12] = (_Float16)r3.x; a[13] = (_Float16)r3.y; a[14] = (_Float16)r3.z; a[15] = (_Float16)r3.w;
    if (kt + 1 < 16)
      __builtin_prefetch(&xrow[k0 + 32 + kbase], 0, 3);
#pragma unroll
    for (int j = 0; j < 8; ++j) {
      const int c = (ntb + j) * 16 + ln16;
      // B[K=k][N=c] = Ahat[c][k]: two 8-half runs per lane matching A (16B aligned)
      const v16h bf = cat8(*(const v8h*)(Ah + (size_t)c * D_ + k0 + kbase), *(const v8h*)(Ah + (size_t)c * D_ + k0 + 16 + kbase));
      acc[j] = __builtin_amdgcn_wmma_f32_16x16x32_f16(false, a, false, bf,
                                                      (short)0, acc[j], false, false);
    }
  }

  // complete row sum-of-squares: lanes L and L^16 hold complementary K halves of row ln16
  const float fullsq = sq + __shfl_xor(sq, 16, 32);
  const float wv = wp[0], bv = bp[0];
  float lsum = 0.f;
  const int mo = (lane < 16) ? 0 : 8;
#pragma unroll
  for (int j = 0; j < 8; ++j) {
    const int c = (ntb + j) * 16 + ln16;
    const float act = active[c];
#pragma unroll
    for (int v = 0; v < 8; ++v) {
      const int r = v + mo;                                // row within this mtile
      const float ns = __shfl(fullsq, r, 32);              // lane r holds row r's ||x||^2
      const float denom = fmaxf(sqrtf(ns), 1e-8f);         // anchors pre-normalized to unit
      const float z = (acc[j][v] / denom) * wv + bv;
      const float l = (float)label[(size_t)(m0 + mtile * 16 + r) * C_ + c];
      // log sigmoid(z) = min(z,0) - log1p(exp(-|z|)); logsig(-z) = logsig(z) - z
      const float ls = fminf(z, 0.f) - log1pf(__expf(-fabsf(z)));
      lsum += (-(ls - (1.f - l) * z)) * act;
    }
  }
  for (int off = 16; off > 0; off >>= 1) lsum += __shfl_xor(lsum, off, 32);
  __shared__ float red[8];
  if (lane == 0) red[wave] = lsum;
  __syncthreads();
  if (threadIdx.x == 0) {
    float s = 0.f;
    for (int i = 0; i < 8; ++i) s += red[i];
    atomicAdd(&misc[1], s);
  }
}

// ---------------- final scalar ----------------
__global__ void k_final(const float* __restrict__ misc, float* __restrict__ out) {
  if (threadIdx.x == 0 && blockIdx.x == 0) {
    const float na = fmaxf(misc[0], 1.f);
    out[0] = misc[1] / ((float)N_ * na);
  }
}

extern "C" void kernel_launch(void* const* d_in, const int* in_sizes, int n_in,
                              void* d_out, int out_size, void* d_ws, size_t ws_size,
                              hipStream_t stream) {
  (void)in_sizes; (void)n_in; (void)out_size; (void)ws_size;
  const float* x     = (const float*)d_in[0];
  const int*   label = (const int*)d_in[1];
  const float* w     = (const float*)d_in[2];
  const float* b     = (const float*)d_in[3];

  float* ws     = (float*)d_ws;
  float* P      = ws + WS_P / 4;
  float* counts = ws + WS_COUNTS / 4;
  float* active = ws + WS_ACTIVE / 4;
  float* misc   = ws + WS_MISC / 4;
  _Float16* Ah  = (_Float16*)((char*)d_ws + WS_AH);
  float* out    = (float*)d_out;

  k_init<<<(WS_ZERO_FLOATS + 255) / 256, 256, 0, stream>>>(ws);
  k_counts<<<N_ / 256, 256, 0, stream>>>(label, counts);
  dim3 g1(64, 4, 4);
  k_gemm1<<<g1, 256, 0, stream>>>(x, label, P);
  k_finalize<<<C_, 256, 0, stream>>>(P, counts, active, misc, Ah);
  k_gemm2_loss<<<N_ / 64, 256, 0, stream>>>(x, label, Ah, active, w, b, misc);
  k_final<<<1, 1, 0, stream>>>(misc, out);
}
